// EdgeDecoder_62045097558106
// MI455X (gfx1250) — matrix operand order repaired and checked by reading, not compiled, not run on hardware
//
#include <hip/hip_runtime.h>
#include <hip/hip_bf16.h>

// EdgeDecoder: out = sigmoid( relu( (left[i]*right[j]) @ W1 + b1 ) @ W2 + b2 )
// D_IN = D_HID = 128, D_OUT = 1, E = 1,000,000.
// f16 WMMA (16x16x32); W1 pre-swizzled in LDS as B fragments.
// Each wave processes M=32 edges (two A tiles) so every B fragment feeds two
// WMMAs; B loads are explicitly double-buffered to hide LDS latency.

#define D_DIM 128

typedef __attribute__((ext_vector_type(16))) _Float16 v16h;
typedef __attribute__((ext_vector_type(8)))  float    v8f;

union FragAB {
    v16h v;
    uint4 q[2];
};

__device__ __forceinline__ float sigmoidf_fast(float x) {
    return 1.0f / (1.0f + __expf(-x));
}

__global__ __launch_bounds__(128)
void edge_decoder_wmma(const float* __restrict__ left,
                       const float* __restrict__ right,
                       const int*   __restrict__ pr0,
                       const int*   __restrict__ pr1,
                       const float* __restrict__ W1,
                       const float* __restrict__ b1,
                       const float* __restrict__ W2,
                       const float* __restrict__ b2,
                       float*       __restrict__ out,
                       int E)
{
    // B fragments of W1 in f16, pre-swizzled into WMMA B layout:
    //   index = ((kc*8 + nt)*32 + lane)*16 + h
    //   half h of lane l: K = kc*32 + (l<16 ? 0 : 16) + h ; N = nt*16 + (l&15)
    __shared__ _Float16 sB[4 * 8 * 32 * 16];      // 32 KB
    __shared__ _Float16 sx[4][32][D_DIM];         // 32 KB : per-wave x (32 edges x 128)
    __shared__ float    sb1[D_DIM];
    __shared__ float    sw2[D_DIM];

    const int tid  = threadIdx.x;
    const int lane = tid & 31;
    const int w    = tid >> 5;           // wave id in block (0..3)
    const int mrow = lane & 15;          // M row / N column within 16
    const int hiK  = (lane & 16) ? 16 : 0;

    // ---- Stage W1 (f32 -> f16, swizzled), b1, W2 ----
    for (int idx = tid; idx < 4 * 8 * 32 * 16; idx += 128) {
        int h  = idx & 15;
        int ln = (idx >> 4) & 31;
        int nt = (idx >> 9) & 7;
        int kc = idx >> 12;
        int K  = kc * 32 + ((ln & 16) ? 16 : 0) + h;
        int N  = nt * 16 + (ln & 15);
        sB[idx] = (_Float16)W1[K * D_DIM + N];
    }
    if (tid < D_DIM) {
        sb1[tid] = b1[tid];
        sw2[tid] = W2[tid];   // W2 is [128,1]
    }
    __syncthreads();

    // ---- Hoist loop-invariant per-lane bias / W2 values ----
    float bvals[8], w2vals[8];
    #pragma unroll
    for (int nt = 0; nt < 8; ++nt) {
        bvals[nt]  = sb1[nt * 16 + mrow];
        w2vals[nt] = sw2[nt * 16 + mrow];
    }

    const float bias2 = b2[0];
    const int nSuper = (E + 127) >> 7;   // 128 edges per block iteration

    for (int s = blockIdx.x; s < nSuper; s += gridDim.x) {
        const int base = s * 128 + w * 32;   // this wave's 32-edge tile

        // Prefetch next iteration's pair indices (global_prefetch_b8 path).
        {
            const int snext = s + gridDim.x;
            if (snext < nSuper) {
                const int nb = snext * 128 + w * 32 + lane;
                __builtin_prefetch(&pr0[nb], 0, 0);
                __builtin_prefetch(&pr1[nb], 0, 0);
            }
        }

        // ---- Gather: x[e][:] = left[i_e][:] * right[j_e][:] -> f16 in LDS ----
        int ei = base + lane;
        if (ei >= E) ei = E - 1;
        const int ia = pr0[ei];
        const int ja = pr1[ei];

        #pragma unroll
        for (int e = 0; e < 32; ++e) {
            const int ii = __shfl(ia, e, 32);
            const int jj = __shfl(ja, e, 32);
            const float4 va = *(const float4*)(left  + ii * D_DIM + lane * 4);
            const float4 vb = *(const float4*)(right + jj * D_DIM + lane * 4);
            union { _Float16 h[4]; uint2 u; } t;
            t.h[0] = (_Float16)(va.x * vb.x);
            t.h[1] = (_Float16)(va.y * vb.y);
            t.h[2] = (_Float16)(va.z * vb.z);
            t.h[3] = (_Float16)(va.w * vb.w);
            *(uint2*)(&sx[w][e][lane * 4]) = t.u;
        }
        __syncthreads();

        const _Float16* xrow0 = &sx[w][mrow][0];        // M-tile 0: edges base+0..15
        const _Float16* xrow1 = &sx[w][16 + mrow][0];   // M-tile 1: edges base+16..31

        float pacc0[8], pacc1[8];
        #pragma unroll
        for (int r = 0; r < 8; ++r) { pacc0[r] = 0.0f; pacc1[r] = 0.0f; }

        // ---- Layer 1: 2 N-groups x 4 kc x 4 nt; each B feeds 2 WMMAs (M0,M1).
        //      B double-buffered: batch n+1 loaded before batch n's WMMAs. ----
        #pragma unroll
        for (int g = 0; g < 2; ++g) {
            v8f acc0[4], acc1[4];
            #pragma unroll
            for (int j = 0; j < 4; ++j) {
                acc0[j] = (v8f){ 0.f, 0.f, 0.f, 0.f, 0.f, 0.f, 0.f, 0.f };
                acc1[j] = (v8f){ 0.f, 0.f, 0.f, 0.f, 0.f, 0.f, 0.f, 0.f };
            }

            FragAB Bbuf[2];
            {   // preload first B fragment of this group
                const _Float16* bp = &sB[(((0 * 8 + g * 4 + 0) * 32) + lane) * 16];
                Bbuf[0].q[0] = *(const uint4*)(bp);
                Bbuf[0].q[1] = *(const uint4*)(bp + 8);
            }

            #pragma unroll
            for (int kc = 0; kc < 4; ++kc) {
                FragAB A0, A1;
                A0.q[0] = *(const uint4*)(xrow0 + kc * 32 + hiK);
                A0.q[1] = *(const uint4*)(xrow0 + kc * 32 + 8 + hiK);
                A1.q[0] = *(const uint4*)(xrow1 + kc * 32 + hiK);
                A1.q[1] = *(const uint4*)(xrow1 + kc * 32 + 8 + hiK);

                #pragma unroll
                for (int j = 0; j < 4; ++j) {
                    const int step = kc * 4 + j;
                    const int cur  = step & 1;
                    // Prefetch next step's B fragment into the other buffer.
                    const int nstep = step + 1;
                    if (nstep < 16) {
                        const int nkc = nstep >> 2;
                        const int nnt = g * 4 + (nstep & 3);
                        const _Float16* bp =
                            &sB[(((nkc * 8 + nnt) * 32) + lane) * 16];
                        Bbuf[cur ^ 1].q[0] = *(const uint4*)(bp);
                        Bbuf[cur ^ 1].q[1] = *(const uint4*)(bp + 8);
                    }
                    acc0[j] = __builtin_amdgcn_wmma_f32_16x16x32_f16(
                                false, A0.v, false, Bbuf[cur].v,
                                (short)0, acc0[j], false, false);
                    acc1[j] = __builtin_amdgcn_wmma_f32_16x16x32_f16(
                                false, A1.v, false, Bbuf[cur].v,
                                (short)0, acc1[j], false, false);
                }
            }

            // Epilogue for this group: bias + ReLU + layer-2 partial dot.
            #pragma unroll
            for (int j = 0; j < 4; ++j) {
                const int nt = g * 4 + j;
                const float bv  = bvals[nt];
                const float w2v = w2vals[nt];
                #pragma unroll
                for (int r = 0; r < 8; ++r) {
                    pacc0[r] = fmaf(fmaxf(acc0[j][r] + bv, 0.0f), w2v, pacc0[r]);
                    pacc1[r] = fmaf(fmaxf(acc1[j][r] + bv, 0.0f), w2v, pacc1[r]);
                }
            }
        }

        // ---- Reduce across the 16-lane half (N columns) ----
        #pragma unroll
        for (int r = 0; r < 8; ++r) {
            float v0 = pacc0[r], v1 = pacc1[r];
            v0 += __shfl_xor(v0, 1, 32);  v1 += __shfl_xor(v1, 1, 32);
            v0 += __shfl_xor(v0, 2, 32);  v1 += __shfl_xor(v1, 2, 32);
            v0 += __shfl_xor(v0, 4, 32);  v1 += __shfl_xor(v1, 4, 32);
            v0 += __shfl_xor(v0, 8, 32);  v1 += __shfl_xor(v1, 8, 32);
            pacc0[r] = v0; pacc1[r] = v1;
        }

        // ---- Sigmoid + store: lane 0 -> M=0..7, lane 16 -> M=8..15 per tile ----
        if (mrow == 0) {
            const int moff = (lane & 16) ? 8 : 0;
            #pragma unroll
            for (int t = 0; t < 2; ++t) {
                const float* p = t ? pacc1 : pacc0;
                const int mb = base + t * 16 + moff;
                float4 o0, o1;
                o0.x = sigmoidf_fast(p[0] + bias2);
                o0.y = sigmoidf_fast(p[1] + bias2);
                o0.z = sigmoidf_fast(p[2] + bias2);
                o0.w = sigmoidf_fast(p[3] + bias2);
                o1.x = sigmoidf_fast(p[4] + bias2);
                o1.y = sigmoidf_fast(p[5] + bias2);
                o1.z = sigmoidf_fast(p[6] + bias2);
                o1.w = sigmoidf_fast(p[7] + bias2);
                if (mb + 7 < E) {
                    *(float4*)(out + mb)     = o0;
                    *(float4*)(out + mb + 4) = o1;
                } else {
                    float tmp[8] = {o0.x,o0.y,o0.z,o0.w,o1.x,o1.y,o1.z,o1.w};
                    for (int r = 0; r < 8; ++r)
                        if (mb + r < E) out[mb + r] = tmp[r];
                }
            }
        }
        __syncthreads();
    }
}

extern "C" void kernel_launch(void* const* d_in, const int* in_sizes, int n_in,
                              void* d_out, int out_size, void* d_ws, size_t ws_size,
                              hipStream_t stream) {
    const float* left  = (const float*)d_in[0];
    const float* right = (const float*)d_in[1];
    const int*   pairs = (const int*)  d_in[2];
    const float* W1    = (const float*)d_in[3];
    const float* b1    = (const float*)d_in[4];
    const float* W2    = (const float*)d_in[5];
    const float* b2    = (const float*)d_in[6];
    float* out = (float*)d_out;

    const int E = in_sizes[2] / 2;           // pairs is [2, E]
    const int* pr0 = pairs;
    const int* pr1 = pairs + E;

    const int nSuper = (E + 127) / 128;
    int blocks = nSuper < 2048 ? nSuper : 2048;
    if (blocks < 1) blocks = 1;

    edge_decoder_wmma<<<blocks, 128, 0, stream>>>(
        left, right, pr0, pr1, W1, b1, W2, b2, out, E);
}
